// TopoLoss_14551349199216
// MI455X (gfx1250) — compile-verified, matrix operand order checked by reading, not run on hardware
//
#include <hip/hip_runtime.h>
#include <hip/hip_bf16.h>
#include <math.h>

// ---------------------------------------------------------------------------
// TopoLoss on MI455X (gfx1250): sublevel 0-dim persistence of two 32x32
// images, diagram matching, matched-cost sum with autoscale.
//
// Design notes (compile-only loop; reasoning from MI455X specs):
//  * Total input = 8KB -> entirely latency bound; 3 small kernels, all
//    combinatorics resident in LDS of a single WGP.
//  * Final reduction uses v_wmma_f32_16x16x32_f16 with a ones A-matrix and
//    hi/lo f16 (Dekker) split of the f32 costs: sum(D) == 16 * sum(B) holds
//    for ANY lane packing of B, so the reduction is layout-agnostic and
//    f32-accurate while running on the CDNA5 matrix pipe.
// ---------------------------------------------------------------------------

typedef __attribute__((ext_vector_type(16))) _Float16 v16h;
typedef __attribute__((ext_vector_type(8)))  float    v8f;

#define IMG_W   32
#define IMG_N   1024          // 32*32
#define MAXP    1024          // max persistence pairs per image (<= N)
#define PADDED  2304          // >= 2*MAXP, multiple of 256 (one WMMA = 256 f32)

struct TopoWS {
  int countA, countB, pad0, pad1;
  int2   pairsA[MAXP];
  int2   pairsB[MAXP];
  float2 diagA[MAXP];
  float2 diagB[MAXP];
  int    matchA[MAXP], matchB[MAXP];
  int    prefA[MAXP],  prefB[MAXP];
  float  costs[PADDED];
};

__device__ __forceinline__ bool kv_less(float va, int ia, float vb, int ib) {
  // stable sort key: (value, index)
  return (va < vb) || (va == vb && ia < ib);
}

__device__ int uf_find(int* par, int i) {
  int r = i;
  while (par[r] != r) r = par[r];
  while (par[i] != r) { int nx = par[i]; par[i] = r; i = nx; }
  return r;
}

// One block per image: bitonic sort in LDS, then single-lane elder-rule
// union-find (exact mirror of the reference), emit (birth,death) pairs and
// their (birth_val, death_val) diagram points.
__global__ void __launch_bounds__(1024)
topo_pairs_kernel(const float* __restrict__ in0,
                  const float* __restrict__ in1,
                  TopoWS* __restrict__ ws) {
  __shared__ float vals[IMG_N];
  __shared__ int   ord[IMG_N];
  __shared__ int   rnk[IMG_N];
  __shared__ int   par[IMG_N];
  __shared__ int   birth[IMG_N];
  __shared__ int   s_np;

  const int tid = threadIdx.x;
  const float* src = (blockIdx.x == 0) ? in0 : in1;          // batch[0] only
  int2*   pairs = (blockIdx.x == 0) ? ws->pairsA : ws->pairsB;
  float2* diag  = (blockIdx.x == 0) ? ws->diagA  : ws->diagB;
  int*    cnt   = (blockIdx.x == 0) ? &ws->countA : &ws->countB;

  __builtin_prefetch(src + tid, 0, 0);   // gfx1250 global_prefetch_b8
  vals[tid] = src[tid];
  ord[tid]  = tid;
  par[tid]  = -1;
  __syncthreads();

  // Bitonic sort of indices by (value, index) -- equals stable argsort.
  for (int k = 2; k <= IMG_N; k <<= 1) {
    for (int j = k >> 1; j > 0; j >>= 1) {
      int ixj = tid ^ j;
      if (ixj > tid) {
        bool up = ((tid & k) == 0);
        int a = ord[tid], b = ord[ixj];
        bool mis = up ? kv_less(vals[b], b, vals[a], a)
                      : kv_less(vals[a], a, vals[b], b);
        if (mis) { ord[tid] = b; ord[ixj] = a; }
      }
      __syncthreads();
    }
  }

  rnk[ord[tid]] = tid;
  __syncthreads();

  // Elder-rule union-find (sequential by construction; 1024 iters in LDS).
  if (tid == 0) {
    int np = 0;
    for (int k = 0; k < IMG_N; ++k) {
      int idx = ord[k];
      int r0 = idx >> 5, c0 = idx & (IMG_W - 1);
      int nb[4]; int nn = 0;
      if (r0 > 0)          nb[nn++] = idx - IMG_W;
      if (r0 < IMG_W - 1)  nb[nn++] = idx + IMG_W;
      if (c0 > 0)          nb[nn++] = idx - 1;
      if (c0 < IMG_W - 1)  nb[nn++] = idx + 1;
      int roots[4]; int nr = 0;
      for (int q = 0; q < nn; ++q) {
        int j = nb[q];
        if (par[j] >= 0) {
          int r = uf_find(par, j);
          bool dup = false;
          for (int t = 0; t < nr; ++t) if (roots[t] == r) dup = true;
          if (!dup) roots[nr++] = r;
        }
      }
      par[idx]   = idx;
      birth[idx] = idx;
      if (nr > 0) {
        int elder = roots[0];
        for (int t = 1; t < nr; ++t)
          if (rnk[birth[roots[t]]] < rnk[birth[elder]]) elder = roots[t];
        for (int t = 0; t < nr; ++t) {
          if (roots[t] != elder) { pairs[np++] = make_int2(birth[roots[t]], idx); }
          par[roots[t]] = elder;
        }
        par[idx] = elder;
      }
    }
    pairs[np++] = make_int2(ord[0], ord[IMG_N - 1]);  // essential class
    *cnt = np;
    s_np = np;
  }
  __syncthreads();

  int np = s_np;
  if (tid < np) {
    int2 p = pairs[tid];
    diag[tid] = make_float2(vals[p.x], vals[p.y]);
  }
}

__device__ __forceinline__ float cheb(float2 a, float2 b) {
  return fmaxf(fabsf(a.x - b.x), fabsf(a.y - b.y));
}

// Deterministic mutual-nearest-neighbor matching on the structured cost
// matrix (real<->real only when it beats both diagonal projections).
__global__ void __launch_bounds__(256)
topo_match_kernel(TopoWS* __restrict__ ws) {
  const int tid = threadIdx.x, bs = blockDim.x;
  const int n = ws->countA, m = ws->countB;

  for (int i = tid; i < MAXP;   i += bs) { ws->matchA[i] = -1; ws->matchB[i] = -1; }
  for (int i = tid; i < PADDED; i += bs) ws->costs[i] = 0.0f;
  __syncthreads();

  for (int it = 0; it < 32; ++it) {
    for (int i = tid; i < n; i += bs) {
      int bj = -1;
      if (ws->matchA[i] < 0) {
        float2 a = ws->diagA[i];
        float halfA = 0.5f * (a.y - a.x);
        float bestSave = 0.0f;
        for (int j = 0; j < m; ++j) {
          if (ws->matchB[j] >= 0) continue;
          float2 b = ws->diagB[j];
          float save = halfA + 0.5f * (b.y - b.x) - cheb(a, b);
          if (save > bestSave) { bestSave = save; bj = j; }
        }
      }
      ws->prefA[i] = bj;
    }
    __syncthreads();
    for (int j = tid; j < m; j += bs) {
      int bi = -1;
      if (ws->matchB[j] < 0) {
        float2 b = ws->diagB[j];
        float halfB = 0.5f * (b.y - b.x);
        float bestSave = 0.0f;
        for (int i = 0; i < n; ++i) {
          if (ws->matchA[i] >= 0) continue;
          float2 a = ws->diagA[i];
          float save = 0.5f * (a.y - a.x) + halfB - cheb(a, b);
          if (save > bestSave) { bestSave = save; bi = i; }
        }
      }
      ws->prefB[j] = bi;
    }
    __syncthreads();
    for (int i = tid; i < n; i += bs) {
      int j = ws->prefA[i];
      if (j >= 0 && ws->prefB[j] == i) { ws->matchA[i] = j; ws->matchB[j] = i; }
    }
    __syncthreads();
  }

  // Matched-cost vector (mirrors C[arange(n+m), cols]): matched -> Chebyshev,
  // unmatched -> half-persistence diagonal cost, dummy-dummy -> 0 (padded).
  for (int i = tid; i < n; i += bs) {
    int j = ws->matchA[i];
    float2 a = ws->diagA[i];
    ws->costs[i] = (j >= 0) ? cheb(a, ws->diagB[j]) : 0.5f * (a.y - a.x);
  }
  for (int j = tid; j < m; j += bs) {
    if (ws->matchB[j] < 0) {
      float2 b = ws->diagB[j];
      ws->costs[n + j] = 0.5f * (b.y - b.x);
    }
  }
}

// One wave (32 lanes, EXEC all-ones as WMMA requires). Sum PADDED f32 values
// on the matrix pipe: A = ones(16x32 f16), B holds hi/lo f16 Dekker-split of
// 256 f32 per call; sum(D) = 16 * sum(hi+lo) = 16 * sum(x) to ~2^-22 rel.
__global__ void __launch_bounds__(32)
topo_reduce_kernel(const TopoWS* __restrict__ ws, float* __restrict__ out) {
  const int lane = threadIdx.x;

  v16h ones;
#pragma unroll
  for (int e = 0; e < 16; ++e) ones[e] = (_Float16)1.0f;

  v8f acc = {};
  for (int c = 0; c < PADDED / 256; ++c) {
    const float* p = ws->costs + c * 256 + lane * 8;
    v16h b;
#pragma unroll
    for (int e = 0; e < 8; ++e) {
      float x = p[e];
      _Float16 hi = (_Float16)x;
      _Float16 lo = (_Float16)(x - (float)hi);
      b[2 * e]     = hi;
      b[2 * e + 1] = lo;
    }
    acc = __builtin_amdgcn_wmma_f32_16x16x32_f16(
        /*neg_a=*/false, ones, /*neg_b=*/false, b,
        /*c_mod=*/(short)0, acc, /*reuse_a=*/false, /*reuse_b=*/false);
  }

  float part = 0.0f;
#pragma unroll
  for (int e = 0; e < 8; ++e) part += acc[e];

  __shared__ float red[32];
  red[lane] = part;
  __syncthreads();
  __builtin_amdgcn_s_wait_tensorcnt(0);   // gfx1250 tensor-counter wait path

  if (lane == 0) {
    float s = 0.0f;
    for (int t = 0; t < 32; ++t) s += red[t];
    s *= (1.0f / 16.0f);                  // ones A replicates each column sum 16x
    // autoscale: d / 10^ceil(log10(d/2)) while d > 2
    float k = 0.0f;
    if (s > 2.0f) k = ceilf(log10f(fmaxf(s, 1e-30f) * 0.5f));
    if (k < 0.0f) k = 0.0f;
    out[0] = s / powf(10.0f, k);
  }
}

extern "C" void kernel_launch(void* const* d_in, const int* in_sizes, int n_in,
                              void* d_out, int out_size, void* d_ws, size_t ws_size,
                              hipStream_t stream) {
  const float* inputs = (const float*)d_in[0];   // (8,32,32) f32, use [0]
  const float* target = (const float*)d_in[1];   // (8,32,32) f32, use [0]
  TopoWS* ws = (TopoWS*)d_ws;                    // ~58 KB scratch
  float* out = (float*)d_out;                    // scalar f32

  topo_pairs_kernel<<<dim3(2), dim3(1024), 0, stream>>>(inputs, target, ws);
  topo_match_kernel<<<dim3(1), dim3(256), 0, stream>>>(ws);
  topo_reduce_kernel<<<dim3(1), dim3(32), 0, stream>>>(ws, out);

  (void)in_sizes; (void)n_in; (void)out_size; (void)ws_size;
}
